// ImprovedGCN_44238163149264
// MI455X (gfx1250) — matrix operand order repaired and checked by reading, compiled, not run on hardware
//
#include <hip/hip_runtime.h>
#include <hip/hip_bf16.h>

// ---------------------------------------------------------------------------
// ImprovedGCN forward for MI455X (gfx1250, wave32).
//
// Roofline: edge gather/scatter (~0.9 GB of L2 traffic per layer; xw/agg are
// 25.6MB each, L2-resident in the 192MB L2) dominates. All dense layers with
// N=64 go through v_wmma_f32_16x16x32_f16 with f32 accumulation, A loaded as
// 2x b128 per lane and B pre-packed into fragment order (2x b128 per lane),
// so the GEMMs are pure memory-bound streams. Conv GEMM epilogue fuses the
// self-loop + bias aggregation init, saving one full 25.6MB pass per layer.
// ---------------------------------------------------------------------------

typedef __attribute__((ext_vector_type(16))) _Float16 v16h;
typedef __attribute__((ext_vector_type(8)))  _Float16 v8h;
typedef __attribute__((ext_vector_type(8)))  float    v8f;

#define THREADS 256
#define GRID(n) ((unsigned)(((n) + THREADS - 1) / THREADS))

// ---------------- elementwise helpers ----------------

__global__ void fill_f32_kernel(float* p, float v, int n) {
    int i = blockIdx.x * blockDim.x + threadIdx.x;
    if (i < n) p[i] = v;
}

__global__ void cvt_f16_kernel(const float* __restrict__ in, _Float16* __restrict__ out, int n) {
    int i = blockIdx.x * blockDim.x + threadIdx.x;
    if (i < n) out[i] = (_Float16)in[i];
}

__global__ void deg_accum_kernel(const int* __restrict__ dst, float* __restrict__ deg, int E) {
    int i = blockIdx.x * blockDim.x + threadIdx.x;
    if (i < E) atomicAdd(&deg[dst[i]], 1.0f);
}

__global__ void rsqrt_inplace_kernel(float* __restrict__ d, int n) {
    int i = blockIdx.x * blockDim.x + threadIdx.x;
    if (i < n) d[i] = rsqrtf(d[i]);
}

// Pack a row-major f32 weight W[K,64] into WMMA-B fragment order (f16):
// bp[(((k0/32)*4 + nt)*32 + lane)*16 + e] = W[k0 + 16*(lane>>4) + e, nt*16 + (lane&15)]
// so each lane's 16-half fragment is one contiguous 32B chunk.
__global__ void pack_b_kernel(const float* __restrict__ w, _Float16* __restrict__ bp, int K) {
    int idx = blockIdx.x * blockDim.x + threadIdx.x;
    if (idx >= K * 64) return;
    int e    = idx & 15;
    int lane = (idx >> 4) & 31;
    int nt   = (idx >> 9) & 3;
    int ks   = idx >> 11;
    int lh   = lane >> 4;
    int n    = nt * 16 + (lane & 15);
    int kk   = ks * 32 + 16 * lh + e;
    bp[idx] = (_Float16)w[(size_t)kk * 64 + n];
}

// ---------------- WMMA GEMM: C[M,64] = A[M,K] @ B[K,64] ---------------------
// Block = 128 threads = 4 waves; each wave computes a 16(M) x 64(N) stripe
// with 4 accumulator tiles. K is constexpr -> k-loop fully unrolled.

template <int K, bool HAS_BIAS, bool RELU, bool WRITE_F16, bool FUSE_AGG>
__global__ void __launch_bounds__(128)
gemm64_wmma_kernel(const _Float16* __restrict__ A, const _Float16* __restrict__ Bp,
                   const float* __restrict__ bias, float* __restrict__ C,
                   _Float16* __restrict__ C16, const float* __restrict__ dinv,
                   const float* __restrict__ aggbias, float* __restrict__ agg, int M)
{
    const int wave = threadIdx.x >> 5;
    const int lane = threadIdx.x & 31;
    const int m0   = blockIdx.x * 64 + wave * 16;
    if (m0 >= M) return;                 // wave-uniform: EXEC stays all-ones

    const int lh  = lane >> 4;           // lane half (0/1)
    const int l15 = lane & 15;

    int am = m0 + l15;                   // A row held by this lane
    if (am >= M) am = M - 1;             // clamp (stores are guarded below)
    const _Float16* Arow = A + (size_t)am * K;

    v8f acc[4] = {v8f{}, v8f{}, v8f{}, v8f{}};

    #pragma unroll
    for (int k0 = 0; k0 < K; k0 += 32) {
        // A fragment (16x32 f16): two contiguous 16B chunks per lane.
        v8h alo = *(const v8h*)(Arow + k0 + 16 * lh);
        v8h ahi = *(const v8h*)(Arow + k0 + 16 * lh + 8);
        v16h a;
        #pragma unroll
        for (int e = 0; e < 8; ++e) { a[e] = alo[e]; a[e + 8] = ahi[e]; }

        #pragma unroll
        for (int nt = 0; nt < 4; ++nt) {
            // Pre-packed B fragment: one contiguous 32B chunk per lane.
            v16h b = *(const v16h*)(Bp + (size_t)(((k0 >> 5) * 4 + nt) * 512 + lane * 16));
            acc[nt] = __builtin_amdgcn_wmma_f32_16x16x32_f16(
                false, a, false, b, (short)0, acc[nt], false, false);
        }
    }

    // C/D layout: VGPR r -> row m0 + r + 8*lh, col = nt*16 + l15
    #pragma unroll
    for (int nt = 0; nt < 4; ++nt) {
        const int n = nt * 16 + l15;
        #pragma unroll
        for (int r = 0; r < 8; ++r) {
            const int m = m0 + r + 8 * lh;
            if (m < M) {
                float v = acc[nt][r];
                if (HAS_BIAS) v += bias[n];
                if (RELU)     v = fmaxf(v, 0.0f);
                C[(size_t)m * 64 + n] = v;
                if (WRITE_F16) C16[(size_t)m * 64 + n] = (_Float16)v;
                if (FUSE_AGG) {                    // agg = xw*dinv^2 + conv bias
                    float di = dinv[m];
                    agg[(size_t)m * 64 + n] = v * di * di + aggbias[n];
                }
            }
        }
    }
}

// ---------------- GCN edge aggregation --------------------------------------

// 16 threads per edge, float4 gather + 4 scalar f32 atomics per thread.
__global__ void edge_scatter_kernel(const int* __restrict__ src, const int* __restrict__ dst,
                                    const float* __restrict__ dinv, const float* __restrict__ xw,
                                    float* __restrict__ agg, int E)
{
    int idx = blockIdx.x * blockDim.x + threadIdx.x;
    int e = idx >> 4;
    if (e >= E) return;
    int q = idx & 15;
    int s = src[e], d = dst[e];
    float w = dinv[s] * dinv[d];
    float4 v = ((const float4*)(xw + (size_t)s * 64))[q];
    float* ad = agg + (size_t)d * 64 + q * 4;
    atomicAdd(ad + 0, v.x * w);
    atomicAdd(ad + 1, v.y * w);
    atomicAdd(ad + 2, v.z * w);
    atomicAdd(ad + 3, v.w * w);
}

// BatchNorm (eval) + residual ReLU + global_add_pool into concat layout.
__global__ void bn_res_pool_kernel(float* __restrict__ h, _Float16* __restrict__ h16,
                                   const float* __restrict__ agg,
                                   const float* __restrict__ gamma, const float* __restrict__ beta,
                                   const float* __restrict__ mean,  const float* __restrict__ var,
                                   const int* __restrict__ batch, float* __restrict__ pooled,
                                   int layer, int n_nodes)
{
    int idx = blockIdx.x * blockDim.x + threadIdx.x;
    if (idx >= n_nodes * 64) return;
    int i = idx >> 6, c = idx & 63;
    float a = agg[idx];
    float y = gamma[c] * (a - mean[c]) * rsqrtf(var[c] + 1e-5f) + beta[c];
    float hv = h[idx] + fmaxf(y, 0.0f);
    h[idx] = hv;
    h16[idx] = (_Float16)hv;
    atomicAdd(&pooled[(size_t)batch[i] * 256 + layer * 64 + c], hv);
}

// ---------------- final head layer (N=10, f32) ------------------------------

__global__ void dense_kernel(const float* __restrict__ in, const float* __restrict__ w,
                             const float* __restrict__ b, float* __restrict__ out,
                             int M, int K, int N, int relu)
{
    int idx = blockIdx.x * blockDim.x + threadIdx.x;
    if (idx >= M * N) return;
    int m = idx / N, n = idx % N;
    float acc = b[n];
    for (int k = 0; k < K; ++k)
        acc = fmaf(in[(size_t)m * K + k], w[(size_t)k * N + n], acc);
    out[idx] = relu ? fmaxf(acc, 0.0f) : acc;
}

// ---------------- launch ----------------------------------------------------

extern "C" void kernel_launch(void* const* d_in, const int* in_sizes, int n_in,
                              void* d_out, int out_size, void* d_ws, size_t ws_size,
                              hipStream_t stream)
{
    (void)in_sizes; (void)n_in; (void)out_size; (void)ws_size;

    const int N = 100000, E = 1200000, G = 4096, H = 64, INCH = 128, OUTD = 10;

    const float* x     = (const float*)d_in[0];
    const int*   src   = (const int*)  d_in[1];
    const int*   dst   = (const int*)  d_in[2];
    const int*   batch = (const int*)  d_in[3];
    const float* w_in  = (const float*)d_in[4];
    const float* b_in  = (const float*)d_in[5];
    const float *w_conv[4], *b_conv[4];
    for (int l = 0; l < 4; ++l) {
        w_conv[l] = (const float*)d_in[6 + 2 * l];
        b_conv[l] = (const float*)d_in[7 + 2 * l];
    }
    const float *bn_g[4], *bn_b[4], *bn_m[4], *bn_v[4];
    for (int l = 0; l < 4; ++l) {
        bn_g[l] = (const float*)d_in[14 + 4 * l + 0];
        bn_b[l] = (const float*)d_in[14 + 4 * l + 1];
        bn_m[l] = (const float*)d_in[14 + 4 * l + 2];
        bn_v[l] = (const float*)d_in[14 + 4 * l + 3];
    }
    const float* w_jk = (const float*)d_in[30];
    const float* b_jk = (const float*)d_in[31];
    const float* w_h1 = (const float*)d_in[32];
    const float* b_h1 = (const float*)d_in[33];
    const float* w_h2 = (const float*)d_in[34];
    const float* b_h2 = (const float*)d_in[35];
    float* out = (float*)d_out;

    // workspace carve-up (256B aligned)
    char* ws = (char*)d_ws;
    size_t off = 0;
    auto take = [&](size_t bytes) -> char* {
        char* p = ws + off;
        off = (off + bytes + 255) & ~(size_t)255;
        return p;
    };
    _Float16* x16      = (_Float16*)take((size_t)N * INCH * 2);
    _Float16* h16      = (_Float16*)take((size_t)N * H * 2);
    float*    h        = (float*)   take((size_t)N * H * 4);
    float*    xw       = (float*)   take((size_t)N * H * 4);
    float*    agg      = (float*)   take((size_t)N * H * 4);
    float*    dinv     = (float*)   take((size_t)N * 4);
    float*    pooled   = (float*)   take((size_t)G * 256 * 4);
    _Float16* pooled16 = (_Float16*)take((size_t)G * 256 * 2);
    float*    z1       = (float*)   take((size_t)G * H * 4);
    _Float16* z1_16    = (_Float16*)take((size_t)G * H * 2);
    float*    z2       = (float*)   take((size_t)G * H * 4);
    _Float16* winP     = (_Float16*)take((size_t)INCH * H * 2);      // packed B frags
    _Float16* wcP      = (_Float16*)take((size_t)4 * H * H * 2);
    _Float16* wjkP     = (_Float16*)take((size_t)256 * H * 2);
    _Float16* wh1P     = (_Float16*)take((size_t)H * H * 2);

    // ---- degrees / norms ----
    fill_f32_kernel<<<GRID(N), THREADS, 0, stream>>>(dinv, 1.0f, N);  // self-loop +1
    deg_accum_kernel<<<GRID(E), THREADS, 0, stream>>>(dst, dinv, E);
    rsqrt_inplace_kernel<<<GRID(N), THREADS, 0, stream>>>(dinv, N);

    // ---- f16 staging: activations + fragment-packed weights ----
    cvt_f16_kernel<<<GRID(N * INCH), THREADS, 0, stream>>>(x, x16, N * INCH);
    pack_b_kernel<<<GRID(INCH * 64), THREADS, 0, stream>>>(w_in, winP, INCH);
    for (int l = 0; l < 4; ++l)
        pack_b_kernel<<<GRID(H * 64), THREADS, 0, stream>>>(w_conv[l], wcP + (size_t)l * H * H, H);
    pack_b_kernel<<<GRID(256 * 64), THREADS, 0, stream>>>(w_jk, wjkP, 256);
    pack_b_kernel<<<GRID(H * 64), THREADS, 0, stream>>>(w_h1, wh1P, H);

    hipMemsetAsync(pooled, 0, (size_t)G * 256 * 4, stream);

    // ---- input projection: h = relu(x @ W_in + b_in) ----
    gemm64_wmma_kernel<128, true, true, true, false><<<(N + 63) / 64, 128, 0, stream>>>(
        x16, winP, b_in, h, h16, nullptr, nullptr, nullptr, N);

    // ---- 4 GCN layers ----
    for (int l = 0; l < 4; ++l) {
        // xw = h @ W_conv ; agg = xw*dinv^2 + b (fused epilogue)
        gemm64_wmma_kernel<64, false, false, false, true><<<(N + 63) / 64, 128, 0, stream>>>(
            h16, wcP + (size_t)l * H * H, nullptr, xw, nullptr, dinv, b_conv[l], agg, N);
        edge_scatter_kernel<<<GRID(E * 16), THREADS, 0, stream>>>(src, dst, dinv, xw, agg, E);
        bn_res_pool_kernel<<<GRID(N * H), THREADS, 0, stream>>>(
            h, h16, agg, bn_g[l], bn_b[l], bn_m[l], bn_v[l], batch, pooled, l, N);
    }

    // ---- JK + MLP head (JK and head1 via WMMA) ----
    cvt_f16_kernel<<<GRID(G * 256), THREADS, 0, stream>>>(pooled, pooled16, G * 256);
    gemm64_wmma_kernel<256, true, true, true, false><<<(G + 63) / 64, 128, 0, stream>>>(
        pooled16, wjkP, b_jk, z1, z1_16, nullptr, nullptr, nullptr, G);
    gemm64_wmma_kernel<64, true, true, false, false><<<(G + 63) / 64, 128, 0, stream>>>(
        z1_16, wh1P, b_h1, z2, nullptr, nullptr, nullptr, nullptr, G);
    dense_kernel<<<GRID(G * OUTD), THREADS, 0, stream>>>(z2, w_h2, b_h2, out, G, H, OUTD, 0);
}
